// SpikingEncoder_15247133901066
// MI455X (gfx1250) — matrix operand order repaired and checked by reading, not compiled, not run on hardware
//
#include <hip/hip_runtime.h>
#include <hip/hip_bf16.h>
#include <math.h>

// ---------------------------------------------------------------------------
// SpikingEncoder on MI455X (gfx1250, wave32, WMMA bf16 16x16x32)
//   B=8, S=1024, V=32000, E=H=512, 4H=2048, L=2
// Strategy:
//   * dense projections -> 128x64-tile WMMA GEMM, double-buffered LDS tiles
//     filled with GLOBAL_LOAD_ASYNC_TO_LDS_B128 (ASYNCcnt), 1 barrier/chunk
//   * 4 sequential SLSTM scans -> persistent 32-WG kernel, Whh slice resident
//     in LDS (gate-interleaved rows, async-staged), 1 WMMA per k-chunk per
//     wave, gate exchange via wave32 shuffles, ping-pong mem broadcast via L2
//     with one device-scope barrier per step (atomic counter + s_sleep spin)
// ---------------------------------------------------------------------------

typedef __attribute__((ext_vector_type(16))) __bf16 v16bf;
typedef __attribute__((ext_vector_type(8)))  float  v8f;

#define B_   8
#define S_   1024
#define H_   512
#define MR   (B_ * S_)      /* 8192 rows of every sequence tensor (row=b*S+s) */
#define G4   (4 * H_)       /* 2048 gate columns */
#define NWG  32             /* persistent workgroups in the recurrence */

__device__ __forceinline__ __bf16 f2bf(float f) {
  unsigned u = __builtin_bit_cast(unsigned, f);
  unsigned r = u + 0x7FFFu + ((u >> 16) & 1u);
  unsigned short h = (unsigned short)(r >> 16);
  return __builtin_bit_cast(__bf16, h);
}
__device__ __forceinline__ float sigmoidf_(float x) { return 1.0f / (1.0f + expf(-x)); }

// --- CDNA5 async global->LDS copy (ASYNCcnt-tracked), 16B per lane ---------
__device__ __forceinline__ void g2l_b128(void* lds, const void* gaddr) {
  const unsigned l = (unsigned)(uintptr_t)lds;   // low 32 bits = LDS offset
  asm volatile("global_load_async_to_lds_b128 %0, %1, off"
               :: "v"(l), "v"(gaddr) : "memory");
}
__device__ __forceinline__ void wait_async0() {
  asm volatile("s_wait_asynccnt 0x0" ::: "memory");
}

struct U32x2 { uint4 lo, hi; };

// CDNA5 A-matrix 16x32 bf16 layout: lanes 0-15 hold K[0..8)+K[16..24),
// lanes 16-31 hold K[8..16)+K[24..32) (ISA 7.12.2).  `rowk` -> element [row,kc].
__device__ __forceinline__ v16bf load_a_frag(const __bf16* rowk, int lane) {
  const int half = (lane >> 4) & 1;
  U32x2 t;
  t.lo = *(const uint4*)(rowk + half * 8);
  t.hi = *(const uint4*)(rowk + 16 + half * 8);
  return __builtin_bit_cast(v16bf, t);
}
// CDNA5 B-matrix 32x16 bf16 layout: lane holds column (lane&15), lanes 0-15 hold
// K[0..8)+K[16..24), lanes 16-31 hold K[8..16)+K[24..32).  `colk` -> element [n, kc] of W (W=B^T).
__device__ __forceinline__ v16bf load_b_frag(const __bf16* colk, int lane) {
  const int half = (lane >> 4) & 1;
  U32x2 t;
  const __bf16* p = colk + half * 8;
  t.lo = *(const uint4*)(p);
  t.hi = *(const uint4*)(p + 16);
  return __builtin_bit_cast(v16bf, t);
}

// ---------------------------------------------------------------------------
// bf16 GEMM:  C[M,N] = A[M,K] * W[N,K]^T (+bias1+bias2), M = MR.
// 128x64 tile / WG, 8 waves (wave = one 16-row m-tile, 4 n-tiles => 4 WMMAs
// per k=32 chunk).  Double-buffered LDS, async global->LDS staging, one
// barrier per chunk, next chunk's copies overlapped with WMMA.
// ---------------------------------------------------------------------------
#define BM 128
#define BN 64
#define BK 32

__global__ void __launch_bounds__(256)
gemm_bf16_kernel(const __bf16* __restrict__ A, int lda,
                 const __bf16* __restrict__ W, int ldw,
                 void* __restrict__ C, int ldc, int c_is_bf16,
                 const float* __restrict__ bias1, const float* __restrict__ bias2,
                 int K) {
  __shared__ __bf16 As[2][BM * BK];   // 8 KB per buffer
  __shared__ __bf16 Ws[2][BN * BK];   // 4 KB per buffer
  const int tid  = threadIdx.x;
  const int lane = tid & 31;
  const int wv   = tid >> 5;                 // 0..7 = m-tile
  const size_t m0 = (size_t)blockIdx.y * BM;
  const size_t n0 = (size_t)blockIdx.x * BN;

  // staging map: thread stages A rows (tid>>2) and (tid>>2)+64, W row (tid>>2),
  // 8 bf16 (16B) at column (tid&3)*8 of the 32-wide chunk.
  const int sr = tid >> 2;                   // 0..63
  const int sc = (tid & 3) << 3;             // 0,8,16,24

  auto stage = [&](int kc, int bb) {
    g2l_b128(&As[bb][sr * BK + sc],        A + (m0 + sr)      * lda + kc + sc);
    g2l_b128(&As[bb][(sr + 64) * BK + sc], A + (m0 + sr + 64) * lda + kc + sc);
    g2l_b128(&Ws[bb][sr * BK + sc],        W + (n0 + sr)      * ldw + kc + sc);
  };

  v8f acc[4];
  #pragma unroll
  for (int t = 0; t < 4; ++t) acc[t] = (v8f){0.f,0.f,0.f,0.f,0.f,0.f,0.f,0.f};

  stage(0, 0);
  const int nk = K / BK;
  for (int i = 0; i < nk; ++i) {
    wait_async0();                 // this wave's copies for chunk i landed
    __syncthreads();               // => everyone's copies landed; prev reads done
    if (i + 1 < nk) stage((i + 1) * BK, (i + 1) & 1);   // overlap with compute
    const __bf16* as = As[i & 1];
    const __bf16* ws = Ws[i & 1];
    const v16bf af = load_a_frag(&as[(wv * 16 + (lane & 15)) * BK], lane);
    #pragma unroll
    for (int t = 0; t < 4; ++t) {
      const v16bf bf = load_b_frag(&ws[(t * 16 + (lane & 15)) * BK], lane);
      acc[t] = __builtin_amdgcn_wmma_f32_16x16x32_bf16(false, af, false, bf, (short)0, acc[t], false, false);
    }
  }

  // C layout: VGPR v, lanes 0-15 -> M=v, N=lane; lanes 16-31 -> M=8+v
  const int mlb = wv * 16 + ((lane >> 4) << 3);
  const int nl  = lane & 15;
  #pragma unroll
  for (int t = 0; t < 4; ++t) {
    const size_t ng = n0 + (size_t)t * 16 + nl;
    float badd = 0.f;
    if (bias1) badd += bias1[ng];
    if (bias2) badd += bias2[ng];
    #pragma unroll
    for (int v = 0; v < 8; ++v) {
      const size_t mg  = m0 + mlb + v;
      const float  val = acc[t][v] + badd;
      if (c_is_bf16) ((__bf16*)C)[mg * ldc + ng] = f2bf(val);
      else           ((float*) C)[mg * ldc + ng] = val;
    }
  }
}

// ---------------------------------------------------------------------------
// Persistent SLSTM recurrence.  32 WGs x 128 thr (4 waves).  Each wave owns a
// 16-row gate-interleaved N-tile (4 h values x {i,f,g,o}); Whh slice (64 rows
// x 512 K, bf16) async-staged into LDS and resident for all 1024 steps.  mem
// is broadcast through a ping-pong global bf16 buffer [2][16][512]; one
// device-scope barrier per step.
// ---------------------------------------------------------------------------
__global__ void __launch_bounds__(128)
slstm_rec_kernel(const float* __restrict__ Xp,        // [MR,2048] x@Wih^T+bih+bhh
                 const __bf16* __restrict__ WhhB,     // [2048,512] bf16
                 const float* __restrict__ thrp,
                 const float* __restrict__ syn0,      // [8,512]
                 __bf16* __restrict__ memG,           // [2][16][512] ping-pong
                 unsigned* __restrict__ counter,
                 __bf16* __restrict__ spk_seq,        // [MR,512] bf16
                 float* __restrict__ synseq,          // [MR,512] or null
                 float* __restrict__ memseq,          // [MR,512] or null
                 float* __restrict__ last_syn,        // [8,512] or null
                 float* __restrict__ last_mem,        // [8,512] or null
                 int flip) {
  __shared__ __bf16 whhS[64 * 512];                   // 64 KB resident slice
  const int tid  = threadIdx.x;
  const int lane = tid & 31;
  const int wv   = tid >> 5;                          // 0..3
  const int wg   = blockIdx.x;                        // 0..31

  // Async-stage gate-interleaved Whh rows: r <-> R=wg*64+r, h=R>>2, g=R&3
  for (int e = tid; e < 64 * 64; e += 128) {
    const int r  = e >> 6;
    const int c8 = (e & 63) << 3;
    const int R  = wg * 64 + r;
    const int h  = R >> 2, g = R & 3;
    g2l_b128(&whhS[r * 512 + c8], WhhB + (size_t)(g * H_ + h) * H_ + c8);
  }
  wait_async0();
  __syncthreads();

  const float thr   = thrp[0];
  const int  Rlane  = (wg * 4 + wv) * 16 + (lane & 15);
  const int  h      = Rlane >> 2;
  const bool active = (lane < 16) && ((lane & 3) == 0); // lane owning gate i of h

  float syn[8];
  #pragma unroll
  for (int v = 0; v < 8; ++v)
    syn[v] = (lane < 16) ? syn0[(size_t)v * H_ + h] : 0.f;

  for (int s = 0; s < S_; ++s) {
    const int srow = flip ? (S_ - 1 - s) : s;
    const __bf16* mg     = memG + (size_t)(s & 1) * 16 * H_;
    __bf16*       mgNext = memG + (size_t)((s + 1) & 1) * 16 * H_;

    // gates_pre = mem @ Whh_slice^T : M=16(8 live), N=16, K=512
    v8f acc = {0.f,0.f,0.f,0.f,0.f,0.f,0.f,0.f};
    const __bf16* arow = mg   + (size_t)(lane & 15) * H_;
    const __bf16* brow = whhS + (size_t)(wv * 16 + (lane & 15)) * H_;
    #pragma unroll 4
    for (int kc = 0; kc < H_; kc += 32) {
      const v16bf af = load_a_frag(arow + kc, lane);
      const v16bf bf = load_b_frag(brow + kc, lane);
      acc = __builtin_amdgcn_wmma_f32_16x16x32_bf16(false, af, false, bf, (short)0, acc, false, false);
    }

    // Prefetch next step's precomputed gate rows (global_prefetch_b8)
    if (s + 1 < S_) {
      const int srn = flip ? (S_ - 2 - s) : (s + 1);
      const void* pf = (const void*)(Xp + ((size_t)(tid & 7) * S_ + srn) * G4 + ((size_t)(tid >> 3) << 7));
      __builtin_prefetch(pf, 0, 1);
    }

    // Gather the 4 gates of this lane's h via wave32 shuffles
    const int sb = lane & ~3;
    float gI[8], gF[8], gG[8], gO[8];
    #pragma unroll
    for (int v = 0; v < 8; ++v) {
      const float a = acc[v];
      gI[v] = __shfl(a, sb + 0, 32);
      gF[v] = __shfl(a, sb + 1, 32);
      gG[v] = __shfl(a, sb + 2, 32);
      gO[v] = __shfl(a, sb + 3, 32);
    }

    if (active) {
      #pragma unroll
      for (int v = 0; v < 8; ++v) {                    // v == batch index b
        const size_t xb = ((size_t)v * S_ + srow) * G4;
        const float i = sigmoidf_(gI[v] + Xp[xb + 0 * H_ + h]);
        const float f = sigmoidf_(gF[v] + Xp[xb + 1 * H_ + h]);
        const float g = tanhf    (gG[v] + Xp[xb + 2 * H_ + h]);
        const float o = sigmoidf_(gO[v] + Xp[xb + 3 * H_ + h]);
        const float sy = f * syn[v] + i * g;
        syn[v] = sy;
        const float me = o * tanhf(sy);
        const float sp = (me - thr) > 0.f ? 1.f : 0.f;
        mgNext[(size_t)v * H_ + h] = f2bf(me);
        const size_t row = (size_t)v * S_ + srow;
        spk_seq[row * H_ + h] = f2bf(sp);
        if (synseq) {
          const size_t rr = (size_t)v * S_ + s;
          synseq[rr * H_ + h] = sy;
          memseq[rr * H_ + h] = me;
        }
        if (last_syn && s == S_ - 1) {
          last_syn[(size_t)v * H_ + h] = sy;
          last_mem[(size_t)v * H_ + h] = me;
        }
      }
    }

    // --- device-scope step barrier (release mem writes, arrive, spin) ---
    __threadfence();
    __syncthreads();
    if (tid == 0) {
      __hip_atomic_fetch_add(counter, 1u, __ATOMIC_ACQ_REL, __HIP_MEMORY_SCOPE_AGENT);
      const unsigned target = (unsigned)(s + 1) * (unsigned)NWG;
      while (__hip_atomic_load(counter, __ATOMIC_ACQUIRE, __HIP_MEMORY_SCOPE_AGENT) < target)
        __builtin_amdgcn_s_sleep(1);
    }
    __syncthreads();
    __threadfence();
  }
}

// ---------------------------------------------------------------------------
// Pointwise helpers
// ---------------------------------------------------------------------------
__global__ void cvt_bf16_kernel(const float* __restrict__ x, __bf16* __restrict__ y, int n) {
  const int i = blockIdx.x * blockDim.x + threadIdx.x;
  if (i < n) y[i] = f2bf(x[i]);
}

__global__ void embed_kernel(const int* __restrict__ src, const float* __restrict__ tab,
                             const float* __restrict__ alpha, const float* __restrict__ w,
                             const float* __restrict__ b, __bf16* __restrict__ out) {
  const int i = blockIdx.x * blockDim.x + threadIdx.x;
  if (i >= MR * H_) return;
  const int r = i >> 9, j = i & 511;
  const float x = tab[(size_t)src[r] * H_ + j];
  out[i] = f2bf(w[j] * tanhf(alpha[0] * x) + b[j]);
}

// DyT on [MR,512] f32 -> bf16 dst (strided; caller pre-offsets dst for concat)
__global__ void dyt_kernel(const float* __restrict__ x, const float* __restrict__ alpha,
                           const float* __restrict__ w, const float* __restrict__ b,
                           __bf16* __restrict__ dst, int ldd) {
  const int i = blockIdx.x * blockDim.x + threadIdx.x;
  if (i >= MR * H_) return;
  const int r = i >> 9, j = i & 511;
  dst[(size_t)r * ldd + j] = f2bf(w[j] * tanhf(alpha[0] * x[i]) + b[j]);
}

__global__ void copy_half_kernel(const __bf16* __restrict__ src, __bf16* __restrict__ dst, int ldd) {
  const int i = blockIdx.x * blockDim.x + threadIdx.x;
  if (i >= MR * H_) return;
  const int r = i >> 9, j = i & 511;
  dst[(size_t)r * ldd + j] = src[i];
}

// out[b,n] = bias[n] + concat(fst,bst)[b,:] . W[n,:]   (tiny: 8x512, K=1024)
__global__ void state_mix_kernel(const float* __restrict__ fst, const float* __restrict__ bst,
                                 const float* __restrict__ W, const float* __restrict__ bias,
                                 float* __restrict__ out) {
  const int i = blockIdx.x * blockDim.x + threadIdx.x;
  if (i >= B_ * H_) return;
  const int b = i >> 9, n = i & 511;
  const float* wr = W + (size_t)n * (2 * H_);
  float acc = bias[n];
  for (int k = 0; k < H_; ++k) acc += fst[b * H_ + k] * wr[k];
  for (int k = 0; k < H_; ++k) acc += bst[b * H_ + k] * wr[H_ + k];
  out[i] = acc;
}

__global__ void prep_kernel(const float* __restrict__ mem_init, const float* __restrict__ syn_init,
                            float* __restrict__ syn0, __bf16* __restrict__ memG,
                            unsigned* __restrict__ counter) {
  const int i = blockIdx.x * blockDim.x + threadIdx.x;
  if (i == 0) *counter = 0u;
  if (i < 2 * 16 * H_)
    memG[i] = (i < B_ * H_ && mem_init) ? f2bf(mem_init[i]) : f2bf(0.f);
  if (i < B_ * H_)
    syn0[i] = syn_init ? syn_init[i] : 0.f;
}

// ---------------------------------------------------------------------------
// Host sequencing (all on `stream`, graph-capture safe)
// ---------------------------------------------------------------------------
extern "C" void kernel_launch(void* const* d_in, const int* in_sizes, int n_in,
                              void* d_out, int out_size, void* d_ws, size_t ws_size,
                              hipStream_t stream) {
  (void)in_sizes; (void)n_in; (void)out_size; (void)ws_size;

  const int*   src  = (const int*)  d_in[0];
  const float* tab  = (const float*)d_in[1];
  const float* e_a  = (const float*)d_in[2];
  const float* e_w  = (const float*)d_in[3];
  const float* e_b  = (const float*)d_in[4];

  struct LP { const float *Wih,*Whh,*bih,*bhh,*thr,*lin_w,*lin_b,*mix_w,*mix_b,*na,*nw,*nb; };
  auto getLP = [&](int base) {
    LP p;
    p.Wih   = (const float*)d_in[base + 0];
    p.Whh   = (const float*)d_in[base + 1];
    p.bih   = (const float*)d_in[base + 2];
    p.bhh   = (const float*)d_in[base + 3];
    p.thr   = (const float*)d_in[base + 4];
    p.lin_w = (const float*)d_in[base + 5];
    p.lin_b = (const float*)d_in[base + 6];
    p.mix_w = (const float*)d_in[base + 7];
    p.mix_b = (const float*)d_in[base + 8];
    p.na    = (const float*)d_in[base + 9];
    p.nw    = (const float*)d_in[base + 10];
    p.nb    = (const float*)d_in[base + 11];
    return p;
  };
  const LP fL = getLP(5), bLr = getLP(17), L0 = getLP(29), L1 = getLP(41);
  const float* syn_w = (const float*)d_in[53];
  const float* syn_b = (const float*)d_in[54];
  const float* mem_w = (const float*)d_in[55];
  const float* mem_b = (const float*)d_in[56];
  const float* outw  = (const float*)d_in[57];
  const float* outb  = (const float*)d_in[58];
  const float* t_na  = (const float*)d_in[59];
  const float* t_nw  = (const float*)d_in[60];
  const float* t_nb  = (const float*)d_in[61];

  // workspace carve (~145 MB)
  char* base = (char*)d_ws;
  size_t off = 0;
  auto carve = [&](size_t bytes) -> char* {
    char* q = base + off; off += (bytes + 255) & ~(size_t)255; return q;
  };
  __bf16* embW   = (__bf16*)carve((size_t)MR * H_ * 2);        // DyT(embed) bf16
  float*  Xproj  = (float*) carve((size_t)MR * G4 * 4);        // x@Wih^T + biases
  __bf16* spkA   = (__bf16*)carve((size_t)MR * H_ * 2);        // spike seq (reused)
  __bf16* mixin  = (__bf16*)carve((size_t)MR * 2 * H_ * 2);    // concat(lin, x)
  float*  mixout = (float*) carve((size_t)MR * H_ * 4);        // f32 GEMM scratch
  __bf16* catb   = (__bf16*)carve((size_t)MR * 2 * H_ * 2);    // concat(f_out, flip(b_out))
  __bf16* layin  = (__bf16*)carve((size_t)MR * H_ * 2);        // current layer input
  __bf16* wslot  = (__bf16*)carve((size_t)G4 * H_ * 2);        // current bf16 weight
  float*  fsyn   = (float*) carve(B_ * H_ * 4);
  float*  fmem   = (float*) carve(B_ * H_ * 4);
  float*  bsyn   = (float*) carve(B_ * H_ * 4);
  float*  bmem   = (float*) carve(B_ * H_ * 4);
  float*  bisyn  = (float*) carve(B_ * H_ * 4);
  float*  bimem  = (float*) carve(B_ * H_ * 4);
  float*  syn0   = (float*) carve(B_ * H_ * 4);
  __bf16* memG   = (__bf16*)carve(2 * 16 * H_ * 2);
  unsigned* ctr  = (unsigned*)carve(256);
  float* out     = (float*)d_out;
  const size_t BSH = (size_t)MR * H_;

  const int PW_GRID = (MR * H_ + 255) / 256;   // pointwise over [MR,512]

  auto cvt = [&](const float* s, int n) {
    cvt_bf16_kernel<<<(n + 255) / 256, 256, 0, stream>>>(s, wslot, n);
  };
  auto gemm = [&](const __bf16* A, int lda, int ldw, void* C, int ldc, int cbf,
                  const float* b1, const float* b2, int N, int K) {
    dim3 g(N / BN, MR / BM);
    gemm_bf16_kernel<<<g, 256, 0, stream>>>(A, lda, wslot, ldw, C, ldc, cbf, b1, b2, K);
  };
  auto prep = [&](const float* mi, const float* si) {
    prep_kernel<<<(2 * 16 * H_ + 255) / 256, 256, 0, stream>>>(mi, si, syn0, memG, ctr);
  };
  auto rec = [&](const LP& L, float* ss, float* ms, float* ls, float* lm, int flip) {
    slstm_rec_kernel<<<NWG, 128, 0, stream>>>(Xproj, wslot, L.thr, syn0, memG, ctr,
                                              spkA, ss, ms, ls, lm, flip);
  };
  auto dyt = [&](const float* alpha, const float* w, const float* b, __bf16* dst, int ldd) {
    dyt_kernel<<<PW_GRID, 256, 0, stream>>>(mixout, alpha, w, b, dst, ldd);
  };

  // 1) embedding gather + encoder DyT -> bf16
  embed_kernel<<<PW_GRID, 256, 0, stream>>>(src, tab, e_a, e_w, e_b, embW);

  // 2) forward branch
  cvt(fL.Wih, G4 * H_);
  gemm(embW, H_, H_, Xproj, G4, 0, fL.bih, fL.bhh, G4, H_);
  cvt(fL.Whh, G4 * H_);
  prep(nullptr, nullptr);
  rec(fL, nullptr, nullptr, fsyn, fmem, /*flip=*/0);
  cvt(fL.lin_w, H_ * H_);
  gemm(spkA, H_, H_, mixin, 2 * H_, 1, fL.lin_b, nullptr, H_, H_);      // lin -> mixin[:, :512]
  copy_half_kernel<<<PW_GRID, 256, 0, stream>>>(embW, mixin + H_, 2 * H_);
  cvt(fL.mix_w, H_ * 2 * H_);
  gemm(mixin, 2 * H_, 2 * H_, mixout, H_, 0, fL.mix_b, nullptr, H_, 2 * H_);
  dyt(fL.na, fL.nw, fL.nb, catb, 2 * H_);                               // f_out -> cat[:, :512]

  // 3) backward branch (flipped indexing; spikes stored pre-flipped)
  cvt(bLr.Wih, G4 * H_);
  gemm(embW, H_, H_, Xproj, G4, 0, bLr.bih, bLr.bhh, G4, H_);
  cvt(bLr.Whh, G4 * H_);
  prep(nullptr, nullptr);
  rec(bLr, nullptr, nullptr, bsyn, bmem, /*flip=*/1);
  cvt(bLr.lin_w, H_ * H_);
  gemm(spkA, H_, H_, mixin, 2 * H_, 1, bLr.lin_b, nullptr, H_, H_);     // emb half still valid
  cvt(bLr.mix_w, H_ * 2 * H_);
  gemm(mixin, 2 * H_, 2 * H_, mixout, H_, 0, bLr.mix_b, nullptr, H_, 2 * H_);
  dyt(bLr.na, bLr.nw, bLr.nb, catb + H_, 2 * H_);                       // flip(b_out) -> cat[:, 512:]

  // 4) state transform + output transform
  state_mix_kernel<<<(B_ * H_ + 255) / 256, 256, 0, stream>>>(fsyn, bsyn, syn_w, syn_b, bisyn);
  state_mix_kernel<<<(B_ * H_ + 255) / 256, 256, 0, stream>>>(fmem, bmem, mem_w, mem_b, bimem);
  cvt(outw, H_ * 2 * H_);
  gemm(catb, 2 * H_, 2 * H_, mixout, H_, 0, outb, nullptr, H_, 2 * H_);
  dyt(t_na, t_nw, t_nb, layin, H_);

  // 5) stacked layers; syn/mem sequences written straight into d_out [4,B,S,H]
  const LP* Ls[2] = { &L0, &L1 };
  for (int li = 0; li < 2; ++li) {
    const LP& L = *Ls[li];
    cvt(L.Wih, G4 * H_);
    gemm(layin, H_, H_, Xproj, G4, 0, L.bih, L.bhh, G4, H_);
    cvt(L.Whh, G4 * H_);
    prep(bimem, bisyn);
    rec(L, out + (size_t)li * BSH, out + (size_t)(2 + li) * BSH, nullptr, nullptr, 0);
    if (li == 0) {   // layer1's post-projection output is unused by the reference
      cvt(L.lin_w, H_ * H_);
      gemm(spkA, H_, H_, mixin, 2 * H_, 1, L.lin_b, nullptr, H_, H_);
      copy_half_kernel<<<PW_GRID, 256, 0, stream>>>(layin, mixin + H_, 2 * H_);
      cvt(L.mix_w, H_ * 2 * H_);
      gemm(mixin, 2 * H_, 2 * H_, mixout, H_, 0, L.mix_b, nullptr, H_, 2 * H_);
      dyt(L.na, L.nw, L.nb, layin, H_);
    }
  }
}
